// Attention_6828998000803
// MI455X (gfx1250) — matrix operand order repaired and checked by reading, not compiled, not run on hardware
//
#include <hip/hip_runtime.h>
#include <math.h>

// Problem constants (from reference)
#define HW     4096     // H*W
#define CIN    128      // C
#define HEADS  4
#define DHEAD  32
#define DIM    128      // HEADS*DHEAD
#define NQKV   384      // 3*DIM
#define SCALEF 10.0f
#define EPSF   1e-8f

typedef __attribute__((ext_vector_type(16))) __bf16 v16bf;
typedef __attribute__((ext_vector_type(8)))  float  v8f;

union BF16x16 { v16bf v; unsigned int u[8]; };
union F32x8   { v8f   v; float        f[8]; };

__device__ __forceinline__ unsigned short f2bf(float f) {
  unsigned int u = __builtin_bit_cast(unsigned int, f);
  u += 0x7FFFu + ((u >> 16) & 1u);            // round-to-nearest-even
  return (unsigned short)(u >> 16);
}

__device__ __forceinline__ unsigned int ld_u32(const void* p) {
  unsigned int v; __builtin_memcpy(&v, p, 4); return v;
}

// reductions across the 16 lanes of each wave32 half (masks < 16 never cross halves)
__device__ __forceinline__ float hmax16(float v) {
  v = fmaxf(v, __shfl_xor(v, 1, 32));
  v = fmaxf(v, __shfl_xor(v, 2, 32));
  v = fmaxf(v, __shfl_xor(v, 4, 32));
  v = fmaxf(v, __shfl_xor(v, 8, 32));
  return v;
}
__device__ __forceinline__ float hsum16(float v) {
  v += __shfl_xor(v, 1, 32);
  v += __shfl_xor(v, 2, 32);
  v += __shfl_xor(v, 4, 32);
  v += __shfl_xor(v, 8, 32);
  return v;
}

// ---------------- f32 -> bf16 conversion ----------------
__global__ void cvt_bf16_kernel(const float* __restrict__ s,
                                unsigned short* __restrict__ d, int n) {
  int i = blockIdx.x * blockDim.x + threadIdx.x;
  if (i < n) d[i] = f2bf(s[i]);
}

// ---------------- generic bf16 WMMA GEMM: C(MxN) = A(MxK) * B(NxK)^T (+bias) ----
// one wave computes one 16x16 tile of C; K multiple of 32, M,N multiples of 16.
__global__ __launch_bounds__(256) void gemm_bf16_kernel(
    const unsigned short* __restrict__ A, const unsigned short* __restrict__ B,
    float* __restrict__ C, int M, int N, int K, const float* __restrict__ bias) {
  int lane = threadIdx.x & 31;
  int w    = (blockIdx.x * blockDim.x + threadIdx.x) >> 5;
  int ntn  = N >> 4;
  int tiles = (M >> 4) * ntn;
  if (w >= tiles) return;
  int tm = (w / ntn) << 4;
  int tn = (w % ntn) << 4;
  int half = lane >> 4, ln = lane & 15;

  F32x8 acc;
#pragma unroll
  for (int g = 0; g < 8; g++) acc.f[g] = 0.0f;

  const unsigned short* arow = A + (size_t)(tm + ln) * K;
  const unsigned short* brow = B + (size_t)(tn + ln) * K;

  for (int k0 = 0; k0 < K; k0 += 32) {
    BF16x16 af, bf;
#pragma unroll
    for (int r = 0; r < 8; r++) {
      // A-frag (16x32 bf16): lane-half interleaved K pairs per ISA table
      int ka = k0 + ((r < 4) ? (2 * r + 8 * half) : (16 + 2 * (r - 4) + 8 * half));
      af.u[r] = ld_u32(arow + ka);
      // B-frag (32x16 bf16): lane-half interleaved K pairs, same order as A
      int kb = k0 + ((r < 4) ? (2 * r + 8 * half) : (16 + 2 * (r - 4) + 8 * half));
      bf.u[r] = ld_u32(brow + kb);
    }
    acc.v = __builtin_amdgcn_wmma_f32_16x16x32_bf16(
        false, af.v, false, bf.v, (short)0, acc.v, false, false);
  }

#pragma unroll
  for (int g = 0; g < 8; g++) {
    int row = tm + g + 8 * half;
    int col = tn + ln;
    float out = acc.f[g];
    if (bias) out += bias[col];
    C[(size_t)row * N + col] = out;
  }
}

// ---------------- column L2 norms over HW for q,k ----------------
// block b in [0,256): b<128 -> q column b ; b>=128 -> k column b-128
__global__ void colnorm_kernel(const float* __restrict__ qkv,
                               float* __restrict__ norms) {
  __shared__ float red[256];
  int t = blockIdx.x;
  int s = t >> 7, c = t & 127;
  const float* p = qkv + s * 128 + c;
  float acc = 0.0f;
  for (int i = threadIdx.x; i < HW; i += 256) {
    float v = p[(size_t)i * NQKV];
    acc += v * v;
  }
  red[threadIdx.x] = acc;
  __syncthreads();
  for (int off = 128; off; off >>= 1) {
    if ((int)threadIdx.x < off) red[threadIdx.x] += red[threadIdx.x + off];
    __syncthreads();
  }
  if (threadIdx.x == 0) norms[t] = sqrtf(red[0]);
}

// ---------------- normalize q,k ; pack q,k head-major bf16 ; pack v transposed ---
__global__ void normpack_kernel(const float* __restrict__ qkv,
                                const float* __restrict__ norms,
                                unsigned short* __restrict__ qn,
                                unsigned short* __restrict__ kn,
                                unsigned short* __restrict__ vt) {
  int idx = blockIdx.x * blockDim.x + threadIdx.x;   // HEADS*HW*DHEAD = 2^19
  if (idx >= HEADS * HW * DHEAD) return;
  int d = idx & 31;
  int i = (idx >> 5) & (HW - 1);
  int h = idx >> 17;
  int col = h * DHEAD + d;
  const float* row = qkv + (size_t)i * NQKV;
  float q = row[col]        / fmaxf(norms[col], EPSF);
  float k = row[128 + col]  / fmaxf(norms[128 + col], EPSF);
  float v = row[256 + col];
  qn[((size_t)h * HW + i) * DHEAD + d] = f2bf(q);
  kn[((size_t)h * HW + i) * DHEAD + d] = f2bf(k);
  vt[((size_t)h * DHEAD + d) * HW + i] = f2bf(v);    // transposed: [h][d][i]
}

// ---------------- flash attention: one wave = 16 query rows of one head -------
// 64-key inner tile: 4 S-WMMAs + 4 PV-WMMAs per step; softmax reduction cost
// (cross-lane shuffles, O rescale) amortized over 64 keys instead of 32.
__global__ __launch_bounds__(256) void flash_kernel(
    const unsigned short* __restrict__ qn, const unsigned short* __restrict__ kn,
    const unsigned short* __restrict__ vt, float* __restrict__ oatt) {
  __shared__ __align__(16) unsigned short Pt[8][16 * 64];   // per-wave P tile (2KB)
  int lane = threadIdx.x & 31;
  int wib  = threadIdx.x >> 5;
  int w    = blockIdx.x * 8 + wib;          // 0..1023
  int h    = w >> 8;                        // 256 query tiles per head
  int it   = w & 255;
  int ibase = it << 4;
  int half = lane >> 4, ln = lane & 15;

  // Q A-fragment (loaded once; K = DHEAD = 32)
  BF16x16 aQ;
  const unsigned short* qrow = qn + ((size_t)h * HW + ibase + ln) * DHEAD;
#pragma unroll
  for (int r = 0; r < 8; r++) {
    int kk = (r < 4) ? (2 * r + 8 * half) : (16 + 2 * (r - 4) + 8 * half);
    aQ.u[r] = ld_u32(qrow + kk);
  }

  float m[8], lsum[8];
  F32x8 O0, O1;
#pragma unroll
  for (int g = 0; g < 8; g++) { m[g] = -1e30f; lsum[g] = 0.0f; O0.f[g] = 0.0f; O1.f[g] = 0.0f; }

  const unsigned short* kbase = kn + (size_t)h * HW * DHEAD;
  const unsigned short* vbase = vt + (size_t)h * DHEAD * HW;
  unsigned short* Pw = &Pt[wib][0];
  v8f zero = {0.f, 0.f, 0.f, 0.f, 0.f, 0.f, 0.f, 0.f};

  for (int j0 = 0; j0 < HW; j0 += 64) {
    if (j0 + 64 < HW) {  // pull next K tile toward the caches
      __builtin_prefetch(kbase + (size_t)(j0 + 64 + ln) * DHEAD, 0, 3);
      __builtin_prefetch(vbase + (size_t)ln * HW + j0 + 64, 0, 3);
    }
    // S = Q * K^T : 4 column tiles of 16 keys each
    F32x8 s[4];
#pragma unroll
    for (int t = 0; t < 4; t++) {
      BF16x16 bk;
#pragma unroll
      for (int r = 0; r < 8; r++) {
        int kk = (r < 4) ? (2 * r + 8 * half) : (16 + 2 * (r - 4) + 8 * half);  // d index
        bk.u[r] = ld_u32(kbase + (size_t)(j0 + 16 * t + ln) * DHEAD + kk);
      }
      s[t].v = __builtin_amdgcn_wmma_f32_16x16x32_bf16(false, aQ.v, false, bk.v,
                                                       (short)0, zero, false, false);
    }
    // online softmax update per C-row (row = g + 8*half, cols striped over 16 lanes)
    float p[4][8];
#pragma unroll
    for (int g = 0; g < 8; g++) {
      float a0 = s[0].f[g] * SCALEF;
      float a1 = s[1].f[g] * SCALEF;
      float a2 = s[2].f[g] * SCALEF;
      float a3 = s[3].f[g] * SCALEF;
      float rm = hmax16(fmaxf(fmaxf(a0, a1), fmaxf(a2, a3)));
      float mn = fmaxf(m[g], rm);
      float alpha = __expf(m[g] - mn);
      float e0 = __expf(a0 - mn);
      float e1 = __expf(a1 - mn);
      float e2 = __expf(a2 - mn);
      float e3 = __expf(a3 - mn);
      float rs = hsum16((e0 + e1) + (e2 + e3));
      lsum[g] = lsum[g] * alpha + rs;
      m[g] = mn;
      O0.f[g] *= alpha;
      O1.f[g] *= alpha;
      p[0][g] = e0; p[1][g] = e1; p[2][g] = e2; p[3][g] = e3;
    }
    // re-layout P: C-format -> LDS (16x64 bf16 row-major) -> A-format
#pragma unroll
    for (int t = 0; t < 4; t++) {
#pragma unroll
      for (int g = 0; g < 8; g++) {
        int row = g + 8 * half;
        Pw[row * 64 + 16 * t + ln] = f2bf(p[t][g]);
      }
    }
    // O += P * V : two K=32 chunks of the 64-key P tile
#pragma unroll
    for (int c = 0; c < 2; c++) {
      BF16x16 aP;
#pragma unroll
      for (int r = 0; r < 8; r++) {
        int kk = 32 * c + ((r < 4) ? (2 * r + 8 * half) : (16 + 2 * (r - 4) + 8 * half));
        aP.u[r] = ld_u32(Pw + ln * 64 + kk);
      }
      BF16x16 bv0, bv1;
#pragma unroll
      for (int r = 0; r < 8; r++) {
        int kk = j0 + 32 * c + ((r < 4) ? (2 * r + 8 * half) : (16 + 2 * (r - 4) + 8 * half));  // j index
        bv0.u[r] = ld_u32(vbase + (size_t)ln * HW + kk);
        bv1.u[r] = ld_u32(vbase + (size_t)(16 + ln) * HW + kk);
      }
      O0.v = __builtin_amdgcn_wmma_f32_16x16x32_bf16(false, aP.v, false, bv0.v,
                                                     (short)0, O0.v, false, false);
      O1.v = __builtin_amdgcn_wmma_f32_16x16x32_bf16(false, aP.v, false, bv1.v,
                                                     (short)0, O1.v, false, false);
    }
  }

#pragma unroll
  for (int g = 0; g < 8; g++) {
    int row = ibase + g + 8 * half;
    float inv = 1.0f / lsum[g];
    oatt[(size_t)row * DIM + h * DHEAD + ln]      = O0.f[g] * inv;
    oatt[(size_t)row * DIM + h * DHEAD + 16 + ln] = O1.f[g] * inv;
  }
}

// ---------------- workspace layout (bytes) ----------------
#define ALIGN256(x) (((x) + 255) & ~((size_t)255))
static const size_t OFF_XB    = 0;                                             // HW*CIN bf16
static const size_t OFF_WIB   = ALIGN256(OFF_XB    + (size_t)HW * CIN * 2);    // NQKV*CIN bf16
static const size_t OFF_WOB   = ALIGN256(OFF_WIB   + (size_t)NQKV * CIN * 2);  // CIN*DIM bf16
static const size_t OFF_QKV   = ALIGN256(OFF_WOB   + (size_t)CIN * DIM * 2);   // HW*NQKV f32
static const size_t OFF_NORMS = ALIGN256(OFF_QKV   + (size_t)HW * NQKV * 4);   // 256 f32
static const size_t OFF_QN    = ALIGN256(OFF_NORMS + 256 * 4);                 // H*HW*DH bf16
static const size_t OFF_KN    = ALIGN256(OFF_QN    + (size_t)HEADS * HW * DHEAD * 2);
static const size_t OFF_VT    = ALIGN256(OFF_KN    + (size_t)HEADS * HW * DHEAD * 2);
static const size_t OFF_OATT  = ALIGN256(OFF_VT    + (size_t)HEADS * HW * DHEAD * 2); // HW*DIM f32
static const size_t OFF_OATTB = ALIGN256(OFF_OATT  + (size_t)HW * DIM * 4);    // HW*DIM bf16

extern "C" void kernel_launch(void* const* d_in, const int* in_sizes, int n_in,
                              void* d_out, int out_size, void* d_ws, size_t ws_size,
                              hipStream_t stream) {
  const float* x     = (const float*)d_in[0];
  const float* w_in  = (const float*)d_in[1];
  const float* w_out = (const float*)d_in[2];
  const float* b_out = (const float*)d_in[3];

  char* ws = (char*)d_ws;
  unsigned short* xb    = (unsigned short*)(ws + OFF_XB);
  unsigned short* wib   = (unsigned short*)(ws + OFF_WIB);
  unsigned short* wob   = (unsigned short*)(ws + OFF_WOB);
  float*          qkv   = (float*)(ws + OFF_QKV);
  float*          norms = (float*)(ws + OFF_NORMS);
  unsigned short* qn    = (unsigned short*)(ws + OFF_QN);
  unsigned short* kn    = (unsigned short*)(ws + OFF_KN);
  unsigned short* vt    = (unsigned short*)(ws + OFF_VT);
  float*          oatt  = (float*)(ws + OFF_OATT);
  unsigned short* oattb = (unsigned short*)(ws + OFF_OATTB);

  // 1) bf16 conversions of inputs
  cvt_bf16_kernel<<<(HW * CIN + 255) / 256, 256, 0, stream>>>(x, xb, HW * CIN);
  cvt_bf16_kernel<<<(NQKV * CIN + 255) / 256, 256, 0, stream>>>(w_in, wib, NQKV * CIN);
  cvt_bf16_kernel<<<(CIN * DIM + 255) / 256, 256, 0, stream>>>(w_out, wob, CIN * DIM);

  // 2) qkv = x @ w_in^T   (M=4096, N=384, K=128) -> 6144 tiles, 8 waves/block
  gemm_bf16_kernel<<<(HW / 16) * (NQKV / 16) / 8, 256, 0, stream>>>(
      xb, wib, qkv, HW, NQKV, CIN, nullptr);

  // 3) column norms of q,k over sequence axis
  colnorm_kernel<<<256, 256, 0, stream>>>(qkv, norms);

  // 4) normalize + pack bf16 (v transposed)
  normpack_kernel<<<(HEADS * HW * DHEAD + 255) / 256, 256, 0, stream>>>(
      qkv, norms, qn, kn, vt);

  // 5) flash attention: 1024 waves = 128 blocks x 8 waves
  flash_kernel<<<128, 256, 0, stream>>>(qn, kn, vt, oatt);

  // 6) output projection: y = oatt @ w_out^T + b_out   (M=4096, N=128, K=128)
  cvt_bf16_kernel<<<(HW * DIM + 255) / 256, 256, 0, stream>>>(oatt, oattb, HW * DIM);
  gemm_bf16_kernel<<<(HW / 16) * (DIM / 16) / 8, 256, 0, stream>>>(
      oattb, wob, (float*)d_out, HW, DIM, CIN, b_out);
}
